// StructureModule_30760555774400
// MI455X (gfx1250) — matrix operand order repaired and checked by reading, not compiled, not run on hardware
//
#include <hip/hip_runtime.h>
#include <hip/hip_bf16.h>

typedef _Float16 h16;
typedef __attribute__((ext_vector_type(16))) _Float16 v16h;
typedef __attribute__((ext_vector_type(8)))  _Float16 v8h;
typedef __attribute__((ext_vector_type(8)))  float    v8f;

// ---------------------------------------------------------------------------
// f32 -> f16 convert (optionally with column padding Ns -> Nd, zero-filled)
// ---------------------------------------------------------------------------
__global__ void k_f32_to_f16(const float* __restrict__ in, h16* __restrict__ out, int n) {
  int i = blockIdx.x * blockDim.x + threadIdx.x;
  if (i < n) out[i] = (h16)in[i];
}

__global__ void k_cvt_pad(const float* __restrict__ in, h16* __restrict__ out,
                          int K, int Ns, int Nd) {
  int i = blockIdx.x * blockDim.x + threadIdx.x;
  if (i < K * Nd) {
    int k = i / Nd, n = i - k * Nd;
    out[i] = (n < Ns) ? (h16)in[k * Ns + n] : (h16)0.f;
  }
}

__global__ void k_pad_vec(const float* __restrict__ in, float* __restrict__ out,
                          int Ns, int Nd) {
  int i = threadIdx.x;
  if (i < Nd) out[i] = (i < Ns) ? in[i] : 0.f;
}

// ---------------------------------------------------------------------------
// LayerNorm over last dim C. One block (128 thr) per row. Optional f32 / f16 out.
// ---------------------------------------------------------------------------
__global__ void k_layernorm(const float* __restrict__ in, const float* __restrict__ g,
                            const float* __restrict__ b, float* __restrict__ outf,
                            h16* __restrict__ outh, int C) {
  long long row = blockIdx.x;
  const float* x = in + row * C;
  __shared__ float red[128];
  int tid = threadIdx.x;
  float s = 0.f;
  for (int c = tid; c < C; c += 128) s += x[c];
  red[tid] = s; __syncthreads();
  for (int off = 64; off > 0; off >>= 1) {
    if (tid < off) red[tid] += red[tid + off];
    __syncthreads();
  }
  float mean = red[0] / (float)C;
  __syncthreads();
  float v = 0.f;
  for (int c = tid; c < C; c += 128) { float d = x[c] - mean; v += d * d; }
  red[tid] = v; __syncthreads();
  for (int off = 64; off > 0; off >>= 1) {
    if (tid < off) red[tid] += red[tid + off];
    __syncthreads();
  }
  float inv = rsqrtf(red[0] / (float)C + 1e-5f);
  for (int c = tid; c < C; c += 128) {
    float y = (x[c] - mean) * inv * g[c] + b[c];
    if (outf) outf[row * C + c] = y;
    if (outh) outh[row * C + c] = (h16)y;
  }
}

// ---------------------------------------------------------------------------
// Batched WMMA GEMM: C[bz] (+)= A[bz] (MxK f16) @ B[bz] (KxN f16) + bias.
// Requirements (guaranteed by host): K%32==0, lda/ldb/sA/sB multiples of 8,
// 16B-aligned bases, B buffers padded so over-reads past N stay in-bounds.
// A fragments: two b128 loads straight from global (CDNA5 16-bit A layout is
// two contiguous 8xf16 runs per lane). B: b128 global load -> LDS transposed
// -> two b128 LDS reads per lane. M/N edges: clamped loads + guarded stores.
// flags: 1=relu, 2=residual add into C.
// ---------------------------------------------------------------------------
struct GemmP {
  const h16* A; long long lda, sA;
  const h16* B; long long ldb, sB;
  float*     C; long long ldc, sC;
  h16*     C16; long long sC16;     // mirror f16 output, same ldc; null ok
  const float* bias;                // per-output-column, null ok
  int M, N, K;
  int flags;
};

__launch_bounds__(256)
__global__ void k_gemm_wmma(GemmP p) {
  __shared__ __align__(16) h16 BsT[8][16 * 32];   // [wave][n][k]
  int w    = threadIdx.x >> 5;
  int lane = threadIdx.x & 31;
  int hh   = lane >> 4;        // lane half
  int l16  = lane & 15;
  int tn = (p.N + 15) >> 4;
  int tm = (p.M + 15) >> 4;
  int tile = blockIdx.x * 8 + w;
  int mt = tile / tn;
  int nt = tile - mt * tn;
  bool valid = (mt < tm);
  if (!valid) { mt = tm - 1; nt = 0; }

  long long bz = blockIdx.z;
  const h16* A = p.A + bz * p.sA;
  const h16* B = p.B + bz * p.sB;
  float*     C = p.C + bz * p.sC;
  h16*     C16 = p.C16 ? (p.C16 + bz * p.sC16) : (h16*)0;

  // A: lane l16 owns matrix row m; clamp for M-edge tiles (stores are guarded).
  int rowA = mt * 16 + l16;
  if (rowA >= p.M) rowA = p.M - 1;
  const h16* Abase = A + (long long)rowA * p.lda;

  // B staging roles: lane -> (k-row, 8-col chunk)
  int bk = lane >> 1;          // 0..15
  int bc = (lane & 1) * 8;     // 0 or 8
  const h16* Bbase = B + nt * 16 + bc;

  v8f acc = {};
  int nk = p.K >> 5;
  for (int kk = 0; kk < nk; ++kk) {
    int k0 = kk << 5;
    // A fragment: two contiguous 8xf16 runs per lane (global b128 loads)
    v8h a0 = *(const v8h*)(Abase + k0 + hh * 8);
    v8h a1 = *(const v8h*)(Abase + k0 + 16 + hh * 8);
    // B tile -> LDS transposed
#pragma unroll
    for (int it = 0; it < 2; ++it) {
      int krow = it * 16 + bk;
      v8h bv = *(const v8h*)(Bbase + (long long)(k0 + krow) * p.ldb);
#pragma unroll
      for (int e = 0; e < 8; ++e) BsT[w][(bc + e) * 32 + krow] = bv[e];
    }
    if (kk + 1 < nk) {
      __builtin_prefetch(Abase + k0 + 32 + hh * 8, 0, 1);
      __builtin_prefetch(Bbase + (long long)(k0 + 32 + bk) * p.ldb, 0, 1);
    }
    __syncthreads();
    v8h b0 = *(const v8h*)(&BsT[w][l16 * 32 + hh * 8]);
    v8h b1 = *(const v8h*)(&BsT[w][l16 * 32 + 16 + hh * 8]);
    v16h a, b;
#pragma unroll
    for (int e = 0; e < 8; ++e) {
      a[e] = a0[e]; a[e + 8] = a1[e];
      b[e] = b0[e]; b[e + 8] = b1[e];
    }
    acc = __builtin_amdgcn_wmma_f32_16x16x32_f16(false, a, false, b,
                                                 (short)0, acc, false, false);
    __syncthreads();
  }

  if (valid) {
#pragma unroll
    for (int r = 0; r < 8; ++r) {
      int row = mt * 16 + r + hh * 8;                 // CDNA5 f32 C/D layout
      int col = nt * 16 + l16;
      if (row < p.M && col < p.N) {
        float v = acc[r];
        if (p.bias) v += p.bias[col];
        if (p.flags & 1) v = v > 0.f ? v : 0.f;
        long long idx = (long long)row * p.ldc + col;
        if (p.flags & 2) v += C[idx];
        C[idx] = v;
        if (C16) C16[idx] = (h16)v;
      }
    }
  }
}

// ---------------------------------------------------------------------------
// q_pts / kv_pts: apply frame (R, t) to linear outputs, scatter to layouts.
// ---------------------------------------------------------------------------
__global__ void k_points(const float* __restrict__ qp, const float* __restrict__ kvp,
                         const float* __restrict__ R, const float* __restrict__ t,
                         float* __restrict__ qpts, float* __restrict__ kpts,
                         h16* __restrict__ vpts16) {
  int i = blockIdx.x;
  int tid = threadIdx.x;     // 0..191
  float Rm[9], tv[3];
  for (int k = 0; k < 9; ++k) Rm[k] = R[i * 9 + k];
  for (int k = 0; k < 3; ++k) tv[k] = t[i * 3 + k];
  if (tid < 48) {
    float x[3];
    for (int d = 0; d < 3; ++d) x[d] = qp[(long long)i * 144 + d * 48 + tid];
    for (int d = 0; d < 3; ++d) {
      float g = Rm[d*3+0]*x[0] + Rm[d*3+1]*x[1] + Rm[d*3+2]*x[2] + tv[d];
      qpts[(long long)i * 144 + tid * 3 + d] = g;
    }
  } else if (tid < 192) {
    int pp = tid - 48;
    float x[3];
    for (int d = 0; d < 3; ++d) x[d] = kvp[(long long)i * 432 + d * 144 + pp];
    int h = pp / 12, pq = pp % 12;
    for (int d = 0; d < 3; ++d) {
      float g = Rm[d*3+0]*x[0] + Rm[d*3+1]*x[1] + Rm[d*3+2]*x[2] + tv[d];
      if (pq < 4) kpts[(long long)i * 144 + (h * 4 + pq) * 3 + d] = g;
      else vpts16[(long long)h * 512 * 24 + (long long)i * 24 + (pq - 4) * 3 + d] = (h16)g;
    }
  }
}

// ---------------------------------------------------------------------------
// Fused attention logits + softmax -> a16[h][i][j]. Block = (i, h), 256 thr.
// bpair has padded stride 16.
// ---------------------------------------------------------------------------
__global__ void k_attn(const float* __restrict__ q, const float* __restrict__ kv,
                       const float* __restrict__ qpts, const float* __restrict__ kpts,
                       const float* __restrict__ bpair, const float* __restrict__ headw,
                       const float* __restrict__ mask, h16* __restrict__ a16) {
  int i = blockIdx.x, h = blockIdx.y;
  int tid = threadIdx.x;
  __shared__ float qs[16], qps[12], red[256];
  __shared__ float sm_hw, sm_mi, smax, ssum;
  if (tid < 16) qs[tid]  = q[(long long)i * 192 + h * 16 + tid];
  if (tid < 12) qps[tid] = qpts[(long long)i * 144 + h * 12 + tid];
  if (tid == 0) {
    float x = headw[h];
    float sp = (x > 20.f) ? x : log1pf(expf(x));   // softplus
    sm_hw = sp * 0.13608276348795434f;             // sqrt(1/54)
    sm_mi = mask[i];
  }
  __syncthreads();
  const float c1 = 0.14433756729740643f;           // sqrt(1/48)
  const float c2 = 0.5773502691896258f;            // sqrt(1/3)
  float lg[2];
#pragma unroll
  for (int u = 0; u < 2; ++u) {
    int j = tid + u * 256;
    const float* kr = kv + (long long)j * 384 + h * 32;
    float dot = 0.f;
#pragma unroll
    for (int c = 0; c < 16; ++c) dot += qs[c] * kr[c];
    const float* kp = kpts + (long long)j * 144 + h * 12;
    float d2 = 0.f;
#pragma unroll
    for (int e = 0; e < 12; ++e) { float dd = qps[e] - kp[e]; d2 += dd * dd; }
    lg[u] = c1 * dot + c2 * bpair[((long long)i * 512 + j) * 16 + h]
          - 0.5f * sm_hw * d2
          + (sm_mi * mask[j] - 1.f) * 100000.f;
  }
  float m = fmaxf(lg[0], lg[1]);
  red[tid] = m; __syncthreads();
  for (int off = 128; off > 0; off >>= 1) {
    if (tid < off) red[tid] = fmaxf(red[tid], red[tid + off]);
    __syncthreads();
  }
  if (tid == 0) smax = red[0];
  __syncthreads();
  float e0 = expf(lg[0] - smax), e1 = expf(lg[1] - smax);
  red[tid] = e0 + e1; __syncthreads();
  for (int off = 128; off > 0; off >>= 1) {
    if (tid < off) red[tid] += red[tid + off];
    __syncthreads();
  }
  if (tid == 0) ssum = red[0];
  __syncthreads();
  float inv = 1.f / ssum;
  h16* arow = a16 + (long long)h * 262144 + (long long)i * 512;
  arow[tid]       = (h16)(e0 * inv);
  arow[tid + 256] = (h16)(e1 * inv);
}

// ---------------------------------------------------------------------------
// Assemble feats (512 x 2112) f16: [o | opt.x | opt.y | opt.z | norm | o_pair]
// ---------------------------------------------------------------------------
__global__ void k_feats(const float* __restrict__ o, const float* __restrict__ opt,
                        const float* __restrict__ opair, const float* __restrict__ R,
                        const float* __restrict__ t, h16* __restrict__ feats) {
  int i = blockIdx.x, tid = threadIdx.x;   // 256 threads
  h16* f = feats + (long long)i * 2112;
  if (tid < 192) {
    int h = tid >> 4, c = tid & 15;
    f[tid] = (h16)o[(long long)h * 8192 + i * 16 + c];
  }
  if (tid < 96) {
    int h = tid >> 3, pz = tid & 7;
    float Rm[9], tv[3];
    for (int k = 0; k < 9; ++k) Rm[k] = R[i * 9 + k];
    for (int k = 0; k < 3; ++k) tv[k] = t[i * 3 + k];
    float g[3], l[3];
    for (int d = 0; d < 3; ++d)
      g[d] = opt[(long long)h * 12288 + (long long)i * 24 + pz * 3 + d] - tv[d];
    for (int d = 0; d < 3; ++d)            // local = R^T (g - t)
      l[d] = Rm[0*3+d]*g[0] + Rm[1*3+d]*g[1] + Rm[2*3+d]*g[2];
    float nrm = sqrtf(l[0]*l[0] + l[1]*l[1] + l[2]*l[2] + 1e-8f);
    f[192 + tid] = (h16)l[0];
    f[288 + tid] = (h16)l[1];
    f[384 + tid] = (h16)l[2];
    f[480 + tid] = (h16)nrm;
  }
  for (int c = tid; c < 1536; c += 256)
    f[576 + c] = (h16)opair[(long long)i * 1536 + c];
}

// ---------------------------------------------------------------------------
// Frames init / quaternion backbone update + outputs (six has stride 16)
// ---------------------------------------------------------------------------
__global__ void k_init_frames(float* R, float* t) {
  int i = blockIdx.x * blockDim.x + threadIdx.x;
  if (i >= 512) return;
  for (int k = 0; k < 9; ++k) R[i * 9 + k] = (k % 4 == 0) ? 1.f : 0.f;
  for (int k = 0; k < 3; ++k) t[i * 3 + k] = 0.f;
}

__global__ void k_frame(const float* __restrict__ six, float* __restrict__ R,
                        float* __restrict__ t, const int* __restrict__ restype,
                        const float* __restrict__ lit, float* __restrict__ out_frames,
                        float* __restrict__ out_pos) {
  int i = blockIdx.x * blockDim.x + threadIdx.x;
  if (i >= 512) return;
  float s0 = six[i*16+0], s1 = six[i*16+1], s2 = six[i*16+2];
  float tx = six[i*16+3], ty = six[i*16+4], tz = six[i*16+5];
  float n = rsqrtf(1.f + s0*s0 + s1*s1 + s2*s2);
  float qw = n, qx = s0*n, qy = s1*n, qz = s2*n;
  float Ru[9];
  Ru[0] = 1.f - 2.f*(qy*qy + qz*qz); Ru[1] = 2.f*(qx*qy - qw*qz); Ru[2] = 2.f*(qx*qz + qw*qy);
  Ru[3] = 2.f*(qx*qy + qw*qz); Ru[4] = 1.f - 2.f*(qx*qx + qz*qz); Ru[5] = 2.f*(qy*qz - qw*qx);
  Ru[6] = 2.f*(qx*qz - qw*qy); Ru[7] = 2.f*(qy*qz + qw*qx); Ru[8] = 1.f - 2.f*(qx*qx + qy*qy);
  float Ro[9], to[3];
  for (int k = 0; k < 9; ++k) Ro[k] = R[i*9+k];
  for (int k = 0; k < 3; ++k) to[k] = t[i*3+k];
  float tn[3];
  for (int d = 0; d < 3; ++d)
    tn[d] = Ro[d*3+0]*tx + Ro[d*3+1]*ty + Ro[d*3+2]*tz + to[d];
  float Rn[9];
  for (int d = 0; d < 3; ++d)
    for (int e = 0; e < 3; ++e) {
      float v = 0.f;
      for (int k = 0; k < 3; ++k) v += Ro[d*3+k] * Ru[k*3+e];
      Rn[d*3+e] = v;
    }
  for (int k = 0; k < 9; ++k) R[i*9+k] = Rn[k];
  for (int k = 0; k < 3; ++k) t[i*3+k] = tn[k];
  float ts0 = tn[0]*10.f, ts1 = tn[1]*10.f, ts2 = tn[2]*10.f;
  float* fo = out_frames + (long long)i * 16;
  fo[0]=Rn[0]; fo[1]=Rn[1]; fo[2]=Rn[2];  fo[3]=ts0;
  fo[4]=Rn[3]; fo[5]=Rn[4]; fo[6]=Rn[5];  fo[7]=ts1;
  fo[8]=Rn[6]; fo[9]=Rn[7]; fo[10]=Rn[8]; fo[11]=ts2;
  fo[12]=0.f;  fo[13]=0.f;  fo[14]=0.f;   fo[15]=1.f;
  int rt = restype[i];
  const float* L = lit + rt * 9;
  float* po = out_pos + (long long)i * 9;
  float tsv[3] = {ts0, ts1, ts2};
  for (int a = 0; a < 3; ++a)
    for (int d = 0; d < 3; ++d)
      po[a*3+d] = Rn[d*3+0]*L[a*3+0] + Rn[d*3+1]*L[a*3+1] + Rn[d*3+2]*L[a*3+2] + tsv[d];
}

// ---------------------------------------------------------------------------
// Host orchestration
// ---------------------------------------------------------------------------
static void gemm(hipStream_t st,
                 const h16* A, long long lda, long long sA,
                 const h16* B, long long ldb, long long sB,
                 float* C, long long ldc, long long sC,
                 h16* C16, long long sC16,
                 const float* bias, int M, int N, int K, int flags, int batch) {
  GemmP p{A, lda, sA, B, ldb, sB, C, ldc, sC, C16, sC16, bias, M, N, K, flags};
  int tiles = ((M + 15) / 16) * ((N + 15) / 16);
  dim3 grid((tiles + 7) / 8, 1, batch);
  k_gemm_wmma<<<grid, 256, 0, st>>>(p);
}

extern "C" void kernel_launch(void* const* d_in, const int* in_sizes, int n_in,
                              void* d_out, int out_size, void* d_ws, size_t ws_size,
                              hipStream_t stream) {
  (void)in_sizes; (void)n_in; (void)out_size; (void)ws_size;
  // Inputs in setup_inputs() dict insertion order (params recursively flattened)
  const float* in_single = (const float*)d_in[0];
  const float* in_pair   = (const float*)d_in[1];
  const float* ln_s_g = (const float*)d_in[2];  const float* ln_s_b = (const float*)d_in[3];
  const float* ln_z_g = (const float*)d_in[4];  const float* ln_z_b = (const float*)d_in[5];
  const float* w_in  = (const float*)d_in[6];   const float* b_in  = (const float*)d_in[7];
  const float* w_q   = (const float*)d_in[8];   const float* b_q   = (const float*)d_in[9];
  const float* w_kv  = (const float*)d_in[10];  const float* b_kv  = (const float*)d_in[11];
  const float* w_qp  = (const float*)d_in[12];  const float* b_qp  = (const float*)d_in[13];
  const float* w_kvp = (const float*)d_in[14];  const float* b_kvp = (const float*)d_in[15];
  const float* w_bz  = (const float*)d_in[16];  const float* b_bz  = (const float*)d_in[17];
  const float* head_w = (const float*)d_in[18];
  const float* w_out = (const float*)d_in[19];  const float* b_out = (const float*)d_in[20];
  const float* ln_i_g = (const float*)d_in[21]; const float* ln_i_b = (const float*)d_in[22];
  const float* w_l1 = (const float*)d_in[23];   const float* b_l1 = (const float*)d_in[24];
  const float* w_l2 = (const float*)d_in[25];   const float* b_l2 = (const float*)d_in[26];
  const float* w_l3 = (const float*)d_in[27];   const float* b_l3 = (const float*)d_in[28];
  const float* ln_t_g = (const float*)d_in[29]; const float* ln_t_b = (const float*)d_in[30];
  const float* w_bb = (const float*)d_in[31];   const float* b_bb = (const float*)d_in[32];
  const float* lit  = (const float*)d_in[33];
  const int*   restype = (const int*)d_in[34];
  const float* mask = (const float*)d_in[35];
  float* out = (float*)d_out;

  // Workspace carve-out (all allocations 256B aligned)
  char* base = (char*)d_ws;
  size_t off = 0;
  auto alloc = [&](size_t bytes) -> void* {
    off = (off + 255) & ~(size_t)255;
    void* p = base + off; off += bytes; return p;
  };
  h16*   z16     = (h16*)alloc(33554432ull * 2);    // LN'd pair, f16 (67MB)
  float* bpair   = (float*)alloc(4194304ull * 4);   // (i,j,16-padded h)
  h16*   a16     = (h16*)alloc(3145728ull * 2);     // (h,i,j)
  float* sbuf    = (float*)alloc(196608ull * 4);
  h16*   s16     = (h16*)alloc(196608ull * 2);
  h16*   x16a    = (h16*)alloc(196608ull * 2);
  float* qbuf    = (float*)alloc(98304ull * 4);     // 512x192
  float* kvbuf   = (float*)alloc(196608ull * 4);
  h16*   kv16    = (h16*)alloc(196608ull * 2);
  float* qp_lin  = (float*)alloc(73728ull * 4);     // 512x144
  float* kvp_lin = (float*)alloc(221184ull * 4);    // 512x432
  float* qpts    = (float*)alloc(73728ull * 4);
  float* kpts    = (float*)alloc(73728ull * 4);
  h16*   vpts16  = (h16*)alloc((147456ull + 128) * 2); // (h,j,24) + N-edge pad
  float* o_ws    = (float*)alloc(98304ull * 4);     // (h,i,16)
  float* opt_ws  = (float*)alloc(147456ull * 4);    // (h,i,24)
  float* opair   = (float*)alloc(786432ull * 4);    // (i,1536)
  h16*   feats16 = (h16*)alloc(1081344ull * 2);     // 512x2112
  float* hA      = (float*)alloc(196608ull * 4);
  h16*   hA16    = (h16*)alloc(196608ull * 2);
  float* hB      = (float*)alloc(196608ull * 4);
  h16*   hB16    = (h16*)alloc(196608ull * 2);
  float* sixb    = (float*)alloc(8192ull * 4);      // 512x16 (padded)
  float* Rws     = (float*)alloc(4608ull * 4);
  float* tws     = (float*)alloc(1536ull * 4);
  h16* w_in16  = (h16*)alloc(147456ull * 2);
  h16* w_q16   = (h16*)alloc(73728ull * 2);
  h16* w_kv16  = (h16*)alloc(147456ull * 2);
  h16* w_qp16  = (h16*)alloc(55296ull * 2);
  h16* w_kvp16 = (h16*)alloc(165888ull * 2);
  h16* w_bz16  = (h16*)alloc(2048ull * 2);          // 128x16 padded
  h16* w_out16 = (h16*)alloc(811008ull * 2);
  h16* w_l116  = (h16*)alloc(147456ull * 2);
  h16* w_l216  = (h16*)alloc(147456ull * 2);
  h16* w_l316  = (h16*)alloc(147456ull * 2);
  h16* w_bb16  = (h16*)alloc(6144ull * 2);          // 384x16 padded
  float* b_bzp = (float*)alloc(16ull * 4);
  float* b_bbp = (float*)alloc(16ull * 4);

  auto cvt = [&](const float* src, h16* dst, int n) {
    k_f32_to_f16<<<(n + 255) / 256, 256, 0, stream>>>(src, dst, n);
  };
  cvt(w_in, w_in16, 147456);   cvt(w_q, w_q16, 73728);
  cvt(w_kv, w_kv16, 147456);   cvt(w_qp, w_qp16, 55296);
  cvt(w_kvp, w_kvp16, 165888); cvt(w_out, w_out16, 811008);
  cvt(w_l1, w_l116, 147456);   cvt(w_l2, w_l216, 147456);
  cvt(w_l3, w_l316, 147456);
  k_cvt_pad<<<(128 * 16 + 255) / 256, 256, 0, stream>>>(w_bz, w_bz16, 128, 12, 16);
  k_cvt_pad<<<(384 * 16 + 255) / 256, 256, 0, stream>>>(w_bb, w_bb16, 384, 6, 16);
  k_pad_vec<<<1, 16, 0, stream>>>(b_bz, b_bzp, 12, 16);
  k_pad_vec<<<1, 16, 0, stream>>>(b_bb, b_bbp, 6, 16);

  // z = LN(pair) -> f16 (once); pair-bias GEMM (once, N padded to 16)
  k_layernorm<<<262144, 128, 0, stream>>>(in_pair, ln_z_g, ln_z_b, nullptr, z16, 128);
  gemm(stream, z16, 128, 0, w_bz16, 16, 0, bpair, 16, 0, nullptr, 0,
       b_bzp, 262144, 16, 128, 0, 1);

  // s = linear_in(LN(single))
  k_layernorm<<<512, 128, 0, stream>>>(in_single, ln_s_g, ln_s_b, nullptr, x16a, 384);
  gemm(stream, x16a, 384, 0, w_in16, 384, 0, sbuf, 384, 0, s16, 0,
       b_in, 512, 384, 384, 0, 1);

  k_init_frames<<<2, 256, 0, stream>>>(Rws, tws);

  for (int blk = 0; blk < 8; ++blk) {
    // IPA linears
    gemm(stream, s16, 384, 0, w_q16, 192, 0, qbuf, 192, 0, nullptr, 0,
         b_q, 512, 192, 384, 0, 1);
    gemm(stream, s16, 384, 0, w_kv16, 384, 0, kvbuf, 384, 0, kv16, 0,
         b_kv, 512, 384, 384, 0, 1);
    gemm(stream, s16, 384, 0, w_qp16, 144, 0, qp_lin, 144, 0, nullptr, 0,
         b_qp, 512, 144, 384, 0, 1);
    gemm(stream, s16, 384, 0, w_kvp16, 432, 0, kvp_lin, 432, 0, nullptr, 0,
         b_kvp, 512, 432, 384, 0, 1);
    k_points<<<512, 192, 0, stream>>>(qp_lin, kvp_lin, Rws, tws, qpts, kpts, vpts16);

    // attention logits + softmax -> a16
    k_attn<<<dim3(512, 12), 256, 0, stream>>>(qbuf, kvbuf, qpts, kpts, bpair,
                                              head_w, mask, a16);

    // o = a @ v          (batch over heads; v = cols 16..31 of each head slab)
    gemm(stream, a16, 512, 262144, kv16 + 16, 384, 32, o_ws, 16, 8192,
         nullptr, 0, nullptr, 512, 16, 512, 0, 12);
    // o_pt = a @ v_pts   (batch over heads)
    gemm(stream, a16, 512, 262144, vpts16, 24, 12288, opt_ws, 24, 12288,
         nullptr, 0, nullptr, 512, 24, 512, 0, 12);
    // o_pair[i] = a[:,i,:] @ z[i]   (batch over residues i; M=12 edge-clamped)
    gemm(stream, a16, 262144, 512, z16, 128, 65536, opair, 128, 1536,
         nullptr, 0, nullptr, 12, 128, 512, 0, 512);

    k_feats<<<512, 256, 0, stream>>>(o_ws, opt_ws, opair, Rws, tws, feats16);

    // s += out_proj(feats)
    gemm(stream, feats16, 2112, 0, w_out16, 384, 0, sbuf, 384, 0, nullptr, 0,
         b_out, 512, 384, 2112, 2, 1);
    k_layernorm<<<512, 128, 0, stream>>>(sbuf, ln_i_g, ln_i_b, sbuf, s16, 384);

    // transition MLP
    gemm(stream, s16, 384, 0, w_l116, 384, 0, hA, 384, 0, hA16, 0,
         b_l1, 512, 384, 384, 1, 1);
    gemm(stream, hA16, 384, 0, w_l216, 384, 0, hB, 384, 0, hB16, 0,
         b_l2, 512, 384, 384, 1, 1);
    gemm(stream, hB16, 384, 0, w_l316, 384, 0, sbuf, 384, 0, nullptr, 0,
         b_l3, 512, 384, 384, 2, 1);
    k_layernorm<<<512, 128, 0, stream>>>(sbuf, ln_t_g, ln_t_b, sbuf, s16, 384);

    // backbone update -> frames + positions (N padded to 16)
    gemm(stream, s16, 384, 0, w_bb16, 16, 0, sixb, 16, 0, nullptr, 0,
         b_bbp, 512, 16, 384, 0, 1);
    k_frame<<<2, 256, 0, stream>>>(sixb, Rws, tws, restype, lit,
                                   out + (long long)blk * 8192,
                                   out + 65536 + (long long)blk * 4608);
  }
}
